// MMD_42356967473641
// MI455X (gfx1250) — matrix operand order repaired and checked by reading, not compiled, not run on hardware
//
#include <hip/hip_runtime.h>
#include <math.h>

typedef __bf16 v16bf __attribute__((ext_vector_type(16)));
typedef float  v8f   __attribute__((ext_vector_type(8)));

#define NROWS 4096           // rows per input matrix
#define NTOT  8192           // total Z rows
#define KDIM  768
#define TILE  128
#define KC    32             // f32 columns staged per chunk
#define LSTRB 40             // LDS row stride in bf16 units (80B: 16B-aligned, conflict-free)
#define NBINS 32768
#define HMAX  4096.0f
#define NBLK  2080           // 64*65/2 lower-triangle blocks

// workspace layout (bytes)
#define WS_NORMS 0
#define WS_HIST  (NTOT * 4)
#define WS_COEF  (WS_HIST + NBINS * 4)
#define WS_PART  (WS_COEF + 64)

// ---------------------------------------------------------------- bf16 split helpers
__device__ __forceinline__ unsigned short f32_to_bf16_rne(float x) {
    unsigned u = __float_as_uint(x);
    return (unsigned short)((u + 0x7FFFu + ((u >> 16) & 1u)) >> 16);
}
__device__ __forceinline__ float bf16_to_f32(unsigned short h) {
    return __uint_as_float(((unsigned)h) << 16);
}
// convert float4 -> 4 hi-bf16 + 4 lo-bf16, store packed (8B each)
__device__ __forceinline__ void cvt_store(unsigned short* ph, unsigned short* pl, float4 v) {
    float vv[4] = { v.x, v.y, v.z, v.w };
    unsigned short h[4], l[4];
#pragma unroll
    for (int i = 0; i < 4; ++i) {
        h[i] = f32_to_bf16_rne(vv[i]);
        l[i] = f32_to_bf16_rne(vv[i] - bf16_to_f32(h[i]));
    }
    uint2 hp = { (unsigned)h[0] | ((unsigned)h[1] << 16), (unsigned)h[2] | ((unsigned)h[3] << 16) };
    uint2 lp = { (unsigned)l[0] | ((unsigned)l[1] << 16), (unsigned)l[2] | ((unsigned)l[3] << 16) };
    *(uint2*)ph = hp;
    *(uint2*)pl = lp;
}

// A fragment (16-bit 16x32): lanes0-15: K0-7 then K16-23 ; lanes16-31: K8-15 then K24-31
__device__ __forceinline__ v16bf load_fragA(const unsigned short* row, int sel) {
    union { v16bf v; uint4 q[2]; } f;
    f.q[0] = *(const uint4*)((const char*)row + sel * 16);
    f.q[1] = *(const uint4*)((const char*)row + sel * 16 + 32);
    return f.v;
}
// B fragment (32x16): lanes0-15: K0-7 then K16-23 ; lanes16-31: K8-15 then K24-31
__device__ __forceinline__ v16bf load_fragB(const unsigned short* row, int sel) {
    union { v16bf v; uint4 q[2]; } f;
    f.q[0] = *(const uint4*)((const char*)row + sel * 16);
    f.q[1] = *(const uint4*)((const char*)row + sel * 16 + 32);
    return f.v;
}

__device__ __forceinline__ void decode_tri(int t, int& bi, int& bj) {
    int r = (int)((sqrtf(8.0f * (float)t + 1.0f) - 1.0f) * 0.5f);
    while ((r + 1) * (r + 2) / 2 <= t) ++r;
    while (r * (r + 1) / 2 > t) --r;
    bi = r;
    bj = t - r * (r + 1) / 2;
}

// ---------------------------------------------------------------- utilities
__global__ void zero_u32_kernel(unsigned* p, int n) {
    int i = blockIdx.x * 256 + threadIdx.x;
    if (i < n) p[i] = 0u;
}

// one wave32 per row: n_i = sum(Z[i]^2), exact fp32
__global__ void row_norms_kernel(const float* __restrict__ X,
                                 const float* __restrict__ Y,
                                 float* __restrict__ norms) {
    int wave = threadIdx.x >> 5;
    int lane = threadIdx.x & 31;
    int row  = blockIdx.x * 8 + wave;
    if (row >= NTOT) return;
    const float* src = (row < NROWS) ? X + (size_t)row * KDIM
                                     : Y + (size_t)(row - NROWS) * KDIM;
    float s = 0.f;
    for (int c = lane; c < KDIM; c += 32) { float v = src[c]; s += v * v; }
    for (int off = 16; off; off >>= 1) s += __shfl_xor(s, off, 32);
    if (lane == 0) norms[row] = s;
}

// ---------------------------------------------------------------- GEMM passes (lower triangle)
// MODE 0: weighted histogram of positive squared distances
// MODE 1: exp(-d2*coef) block sums -> partials (weighted at final reduce)
template <int MODE>
__global__ __launch_bounds__(256)
void gemm_pass_kernel(const float* __restrict__ X, const float* __restrict__ Y,
                      const float* __restrict__ norms,
                      unsigned* __restrict__ ghist,
                      const float* __restrict__ coefp,
                      float* __restrict__ partials) {
    constexpr int TBY = TILE * LSTRB * 2;                 // one bf16 tile: 10240 B
    constexpr int TILE_BYTES = 4 * TBY;                   // AHI/ALO/BHI/BLO: 40960 B
    constexpr int SMEM_BYTES = (MODE == 0) ? (NBINS * 4)  // hist (131072) overlays tiles
                                           : (TILE_BYTES + 256 * 4);
    __shared__ __align__(16) char smem_raw[SMEM_BYTES];

    unsigned short (*lsAhi)[LSTRB] = (unsigned short (*)[LSTRB])(smem_raw);
    unsigned short (*lsAlo)[LSTRB] = (unsigned short (*)[LSTRB])(smem_raw + TBY);
    unsigned short (*lsBhi)[LSTRB] = (unsigned short (*)[LSTRB])(smem_raw + 2 * TBY);
    unsigned short (*lsBlo)[LSTRB] = (unsigned short (*)[LSTRB])(smem_raw + 3 * TBY);

    const int tid  = threadIdx.x;
    const int lane = tid & 31;
    const int wave = tid >> 5;
    const int wm   = wave >> 1;                 // 0..3 -> 32 rows each
    const int wn   = wave & 1;                  // 0..1 -> 64 cols each

    int bi, bj;
    decode_tri(blockIdx.x, bi, bj);             // lower triangle: bj <= bi
    const int row_base = bi * TILE;
    const int col_base = bj * TILE;

    v8f acc[2][4];
#pragma unroll
    for (int m = 0; m < 2; ++m)
#pragma unroll
        for (int n = 0; n < 4; ++n)
            acc[m][n] = (v8f){0.f, 0.f, 0.f, 0.f, 0.f, 0.f, 0.f, 0.f};

    const int lm   = lane & 15;
    const int lsel = lane >> 4;

    for (int k0 = 0; k0 < KDIM; k0 += KC) {
        __syncthreads();
        // stage 128x32 f32 -> bf16 hi/lo tiles for both row-sets
#pragma unroll
        for (int t = 0; t < 4; ++t) {
            int idx = tid + t * 256;            // 0..1023 float4 slots
            int r = idx >> 3, c = (idx & 7) * 4;
            {
                int ga = row_base + r;
                const float* sa = (ga < NROWS) ? X + (size_t)ga * KDIM
                                               : Y + (size_t)(ga - NROWS) * KDIM;
                float4 v = *(const float4*)(sa + k0 + c);
                cvt_store(&lsAhi[r][c], &lsAlo[r][c], v);
            }
            {
                int gb = col_base + r;
                const float* sb = (gb < NROWS) ? X + (size_t)gb * KDIM
                                               : Y + (size_t)(gb - NROWS) * KDIM;
                float4 v = *(const float4*)(sb + k0 + c);
                cvt_store(&lsBhi[r][c], &lsBlo[r][c], v);
            }
        }
        __syncthreads();

        v16bf ah[2], al[2], bh[4], bl[4];
#pragma unroll
        for (int m = 0; m < 2; ++m) {
            ah[m] = load_fragA(lsAhi[wm * 32 + m * 16 + lm], lsel);
            al[m] = load_fragA(lsAlo[wm * 32 + m * 16 + lm], lsel);
        }
#pragma unroll
        for (int n = 0; n < 4; ++n) {
            bh[n] = load_fragB(lsBhi[wn * 64 + n * 16 + lm], lsel);
            bl[n] = load_fragB(lsBlo[wn * 64 + n * 16 + lm], lsel);
        }
        // bf16x3: hi*hi + hi*lo + lo*hi (lo*lo dropped: ~1e-5 relative)
#pragma unroll
        for (int m = 0; m < 2; ++m)
#pragma unroll
            for (int n = 0; n < 4; ++n) {
                acc[m][n] = __builtin_amdgcn_wmma_f32_16x16x32_bf16(
                    false, ah[m], false, bh[n], (short)0, acc[m][n], false, false);
                acc[m][n] = __builtin_amdgcn_wmma_f32_16x16x32_bf16(
                    false, ah[m], false, bl[n], (short)0, acc[m][n], false, false);
                acc[m][n] = __builtin_amdgcn_wmma_f32_16x16x32_bf16(
                    false, al[m], false, bh[n], (short)0, acc[m][n], false, false);
            }
    }

    // C/D element mapping: lane<16 -> M=r,N=lane ; lane>=16 -> M=r+8,N=lane-16
    const int nl = lane & 15;
    const int mh = (lane >> 4) * 8;
    const unsigned w = (bi == bj) ? 1u : 2u;    // symmetry weight

    if (MODE == 0) {
        unsigned* lhist = (unsigned*)smem_raw;  // reuse tile LDS
        __syncthreads();
        for (int b = tid; b < NBINS; b += 256) lhist[b] = 0u;
        __syncthreads();
        const float scale = (float)NBINS / HMAX;
#pragma unroll
        for (int m = 0; m < 2; ++m)
#pragma unroll
            for (int n = 0; n < 4; ++n) {
                int j = col_base + wn * 64 + n * 16 + nl;
                float nj = norms[j];
#pragma unroll
                for (int r = 0; r < 8; ++r) {
                    int i = row_base + wm * 32 + m * 16 + mh + r;
                    float d2 = norms[i] + nj - 2.f * acc[m][n][r];
                    if (d2 > 0.f) {
                        int b = (int)(d2 * scale);
                        b = (b < NBINS) ? b : (NBINS - 1);
                        atomicAdd(&lhist[b], w);
                    }
                }
            }
        __syncthreads();
        for (int b = tid; b < NBINS; b += 256) {
            unsigned c = lhist[b];
            if (c) atomicAdd(&ghist[b], c);     // integer atomics: deterministic
        }
    } else {
        const float coef = coefp[0];            // 1/(2*sigma^2)
        float s = 0.f;
#pragma unroll
        for (int m = 0; m < 2; ++m)
#pragma unroll
            for (int n = 0; n < 4; ++n) {
                int j = col_base + wn * 64 + n * 16 + nl;
                float nj = norms[j];
#pragma unroll
                for (int r = 0; r < 8; ++r) {
                    int i = row_base + wm * 32 + m * 16 + mh + r;
                    float d2 = norms[i] + nj - 2.f * acc[m][n][r];
                    d2 = fmaxf(d2, 0.f);
                    s += __expf(-d2 * coef);
                }
            }
        float* red = (float*)(smem_raw + TILE_BYTES);
        __syncthreads();
        red[tid] = s;
        __syncthreads();
        for (int st = 128; st; st >>= 1) {
            if (tid < st) red[tid] += red[tid + st];
            __syncthreads();
        }
        if (tid == 0) partials[blockIdx.x] = red[0];
    }
}

// ---------------------------------------------------------------- sigma (median heuristic)
__global__ void find_sigma_kernel(const unsigned* __restrict__ hist,
                                  float* __restrict__ coefp) {
    __shared__ unsigned csum[256];
    __shared__ float    cwsum[256];
    const int tid = threadIdx.x;
    const float width = HMAX / (float)NBINS;
    unsigned s = 0; float ws = 0.f;
    for (int b = tid * (NBINS / 256); b < (tid + 1) * (NBINS / 256); ++b) {
        unsigned c = hist[b];
        s += c;
        ws += (float)c * ((b + 0.5f) * width);
    }
    csum[tid] = s; cwsum[tid] = ws;
    __syncthreads();
    if (tid == 0) {
        unsigned total = 0; float wtot = 0.f;
        for (int c = 0; c < 256; ++c) { total += csum[c]; wtot += cwsum[c]; }
        float med = 0.f;
        if (total > 0) {
            unsigned ks[2] = { (total - 1) >> 1, total >> 1 };  // even-count median
            float v[2];
            for (int t = 0; t < 2; ++t) {
                unsigned target = ks[t];
                unsigned cum = 0; int c = 0;
                while (c < 256 && cum + csum[c] <= target) { cum += csum[c]; ++c; }
                float val = HMAX;
                if (c < 256) {
                    for (int b = c * (NBINS / 256); b < (c + 1) * (NBINS / 256); ++b) {
                        cum += hist[b];
                        if (cum > target) { val = (b + 0.5f) * width; break; }
                    }
                }
                v[t] = val;
            }
            med = 0.5f * (v[0] + v[1]);
        }
        float mean_pos = (total > 0) ? (wtot / (float)total) : 0.f;
        if (med <= 0.f) med = mean_pos;
        med = fmaxf(med, 0.01f);
        float sigma = 0.5f * med;               // SIGMA_SCALE * med
        coefp[0] = 1.0f / (2.0f * sigma * sigma);
    }
}

// ---------------------------------------------------------------- final deterministic reduce
__global__ void final_reduce_kernel(const float* __restrict__ partials,
                                    float* __restrict__ out) {
    __shared__ float sx[256], sy[256], sxy[256];
    const int tid = threadIdx.x;
    float a = 0.f, b = 0.f, c = 0.f;
    for (int p = tid; p < NBLK; p += 256) {
        int bi, bj;
        decode_tri(p, bi, bj);
        float v = partials[p] * ((bi == bj) ? 1.f : 2.f);
        if (bi < 32 && bj < 32) a += v;
        else if (bi >= 32 && bj >= 32) b += v;
        else c += v;                            // covers both XY and YX via weight
    }
    sx[tid] = a; sy[tid] = b; sxy[tid] = c;
    __syncthreads();
    for (int st = 128; st; st >>= 1) {
        if (tid < st) { sx[tid] += sx[tid + st]; sy[tid] += sy[tid + st]; sxy[tid] += sxy[tid + st]; }
        __syncthreads();
    }
    // XX+YY-2*XY = (sXX + sYY - sXY_both)/N^2   (sXY_both spans 2*N^2 entries)
    if (tid == 0) out[0] = (sx[0] + sy[0] - sxy[0]) * (1.0f / (4096.0f * 4096.0f));
}

extern "C" void kernel_launch(void* const* d_in, const int* in_sizes, int n_in,
                              void* d_out, int out_size, void* d_ws, size_t ws_size,
                              hipStream_t stream) {
    const float* X = (const float*)d_in[0];
    const float* Y = (const float*)d_in[1];
    char* ws = (char*)d_ws;
    float*    norms    = (float*)(ws + WS_NORMS);
    unsigned* hist     = (unsigned*)(ws + WS_HIST);
    float*    coefp    = (float*)(ws + WS_COEF);
    float*    partials = (float*)(ws + WS_PART);

    const int zcount = NBINS + 16 + NBLK;       // hist + coef pad + partials (contiguous)
    hipLaunchKernelGGL(zero_u32_kernel, dim3((zcount + 255) / 256), dim3(256), 0, stream,
                       hist, zcount);
    hipLaunchKernelGGL(row_norms_kernel, dim3(NTOT / 8), dim3(256), 0, stream, X, Y, norms);

    dim3 grid(NBLK), blk(256);
    hipLaunchKernelGGL((gemm_pass_kernel<0>), grid, blk, 0, stream,
                       X, Y, norms, hist, coefp, partials);
    hipLaunchKernelGGL(find_sigma_kernel, dim3(1), dim3(256), 0, stream, hist, coefp);
    hipLaunchKernelGGL((gemm_pass_kernel<1>), grid, blk, 0, stream,
                       X, Y, norms, hist, coefp, partials);
    hipLaunchKernelGGL(final_reduce_kernel, dim3(1), dim3(256), 0, stream,
                       partials, (float*)d_out);
}
